// PatchedDeepseekMoE_75058848465334
// MI455X (gfx1250) — matrix operand order repaired and checked by reading, not compiled, not run on hardware
//
#include <hip/hip_runtime.h>

typedef __bf16 bf16_t;
typedef __attribute__((ext_vector_type(16))) __bf16 v16bf;
typedef __attribute__((ext_vector_type(8)))  __bf16 v8bf;
typedef __attribute__((ext_vector_type(8)))  float  v8f;

static constexpr int Tt = 2048;   // tokens
static constexpr int Hh = 2048;   // hidden
static constexpr int Ff = 1408;   // expert intermediate
static constexpr int Fs = 2816;   // shared intermediate
static constexpr int Ee = 8;      // experts

static constexpr int BM = 128, BN = 64, BK = 32;
static constexpr int LDK = BK + 8;   // padded LDS row (16B-aligned chunks, spreads banks)

__global__ void zero_cnt_kernel(int* c) {
    if (threadIdx.x < Ee) c[threadIdx.x] = 0;
}

__global__ void convert_x_kernel(const float* __restrict__ x, bf16_t* __restrict__ xb) {
    int i = (blockIdx.x * blockDim.x + threadIdx.x) * 4;
    float4 v = *(const float4*)(x + i);
    xb[i + 0] = (bf16_t)v.x; xb[i + 1] = (bf16_t)v.y;
    xb[i + 2] = (bf16_t)v.z; xb[i + 3] = (bf16_t)v.w;
}

// One wave32 per token: logits -> softmax -> top-2 -> per-expert token lists.
__global__ void gate_topk_kernel(const float* __restrict__ x, const float* __restrict__ gw,
                                 int* __restrict__ cnt, int* __restrict__ tok,
                                 float* __restrict__ wv) {
    const int wave = threadIdx.x >> 5;
    const int lane = threadIdx.x & 31;
    const int t = blockIdx.x * 8 + wave;
    if (t >= Tt) return;

    float acc[Ee];
#pragma unroll
    for (int e = 0; e < Ee; e++) acc[e] = 0.0f;

    for (int h = lane; h < Hh; h += 32) {
        float xv = x[(size_t)t * Hh + h];
#pragma unroll
        for (int e = 0; e < Ee; e++) acc[e] += xv * gw[(size_t)e * Hh + h];
    }
#pragma unroll
    for (int e = 0; e < Ee; e++) {
        float v = acc[e];
        v += __shfl_xor(v, 16, 32);
        v += __shfl_xor(v, 8, 32);
        v += __shfl_xor(v, 4, 32);
        v += __shfl_xor(v, 2, 32);
        v += __shfl_xor(v, 1, 32);
        acc[e] = v;
    }
    if (lane == 0) {
        float mx = acc[0];
#pragma unroll
        for (int e = 1; e < Ee; e++) mx = fmaxf(mx, acc[e]);
        float p[Ee], se = 0.0f;
#pragma unroll
        for (int e = 0; e < Ee; e++) { p[e] = __expf(acc[e] - mx); se += p[e]; }
        float inv = 1.0f / se;
#pragma unroll
        for (int e = 0; e < Ee; e++) p[e] *= inv;

        int i0 = 0; float s0 = p[0];
#pragma unroll
        for (int e = 1; e < Ee; e++) if (p[e] > s0) { s0 = p[e]; i0 = e; }
        int i1 = -1; float s1 = -1.0f;
#pragma unroll
        for (int e = 0; e < Ee; e++) if (e != i0 && p[e] > s1) { s1 = p[e]; i1 = e; }

        int p0 = atomicAdd(&cnt[i0], 1);
        tok[i0 * Tt + p0] = t; wv[i0 * Tt + p0] = s0;
        int p1 = atomicAdd(&cnt[i1], 1);
        tok[i1 * Tt + p1] = t; wv[i1 * Tt + p1] = s1;
    }
}

__device__ __forceinline__ v16bf cat16(v8bf lo, v8bf hi) {
    return __builtin_shufflevector(lo, hi, 0, 1, 2, 3, 4, 5, 6, 7,
                                   8, 9, 10, 11, 12, 13, 14, 15);
}

// LDS aperture: low 32 bits of the flat pointer are the LDS byte offset.
__device__ __forceinline__ unsigned lds_off(const void* p) {
    return (unsigned)(unsigned long long)p;
}

// Async global->LDS copy of 32 bytes (two b128, INST_OFFSET applies to both sides).
__device__ __forceinline__ void async_copy32(unsigned lds_dst, const void* gsrc) {
    asm volatile("global_load_async_to_lds_b128 %0, %1, off\n\t"
                 "global_load_async_to_lds_b128 %0, %1, off offset:16"
                 :: "v"(lds_dst), "v"(gsrc) : "memory");
}
__device__ __forceinline__ void wait_async() {
    asm volatile("s_wait_asynccnt 0x0" ::: "memory");
}

// GEMM1: act = silu(X@Wg^T) * (X@Wu^T) * rowscale, output bf16.
// ROUTED: per-expert gathered token rows, rowscale = gate prob.
template <bool ROUTED>
__global__ __launch_bounds__(256) void gemm_gu_act_kernel(
    const bf16_t* __restrict__ Xb,   // [Tt, Hh] bf16
    const float* __restrict__ W,     // [(E*)2F, H] fp32, rows [0,F)=gate, [F,2F)=up
    bf16_t* __restrict__ Out,        // [(E*Tt | Tt), F] bf16
    const int* __restrict__ cnt, const int* __restrict__ tok,
    const float* __restrict__ wv, const int F) {
    const int e = blockIdx.z;
    const int nrows = ROUTED ? cnt[e] : Tt;
    const int mbase = blockIdx.y * BM;
    if (mbase >= nrows) return;
    const int nbase = blockIdx.x * BN;
    const float* Wg = W + (ROUTED ? (size_t)e * 2 * F * Hh : 0);
    const float* Wu = Wg + (size_t)F * Hh;

    __shared__ alignas(16) bf16_t As[2][BM][LDK];
    __shared__ alignas(16) bf16_t Bg[2][BN][LDK];
    __shared__ alignas(16) bf16_t Bu[2][BN][LDK];
    __shared__ int rTok[BM];
    __shared__ float rW[BM];

    const int tid = threadIdx.x;
    if (tid < BM) {
        int r = mbase + tid;
        if constexpr (ROUTED) {
            if (r < nrows) { rTok[tid] = tok[e * Tt + r]; rW[tid] = wv[e * Tt + r]; }
            else           { rTok[tid] = 0;              rW[tid] = 0.0f; }
        } else {
            rTok[tid] = r; rW[tid] = 1.0f;
        }
    }
    __syncthreads();

    const int wave = tid >> 5, lane = tid & 31;
    const int wm = wave & 3, wn = wave >> 2;   // 4 x 2 wave grid -> 32x32 per wave
    const int lm = lane & 15, kh = lane >> 4;

    const int arow = tid >> 1, ah = (tid & 1) * 16;   // A: 128 rows x (2 x 16 bf16)
    const int brow = tid >> 2, bq = (tid & 3) * 8;    // B: 64 rows x (4 x 8 f32)

    const bf16_t* aSrc = Xb + (size_t)rTok[arow] * Hh + ah;
    const float*  gSrc = Wg + (size_t)(nbase + brow) * Hh + bq;
    const float*  uSrc = Wu + (size_t)(nbase + brow) * Hh + bq;
    const unsigned aDst0 = lds_off(&As[0][arow][ah]);
    const unsigned aDst1 = lds_off(&As[1][arow][ah]);

    v8f cg[2][2] = {}; v8f cu[2][2] = {};
    float4 rg0, rg1, ru0, ru1;   // register-staged B tile (next buffer)

    auto loadB = [&](int k0) {
        rg0 = *(const float4*)(gSrc + k0);
        rg1 = *(const float4*)(gSrc + k0 + 4);
        ru0 = *(const float4*)(uSrc + k0);
        ru1 = *(const float4*)(uSrc + k0 + 4);
    };
    auto commitB = [&](int buf) {
        bf16_t* dg = &Bg[buf][brow][bq];
        bf16_t* du = &Bu[buf][brow][bq];
        dg[0] = (bf16_t)rg0.x; dg[1] = (bf16_t)rg0.y; dg[2] = (bf16_t)rg0.z; dg[3] = (bf16_t)rg0.w;
        dg[4] = (bf16_t)rg1.x; dg[5] = (bf16_t)rg1.y; dg[6] = (bf16_t)rg1.z; dg[7] = (bf16_t)rg1.w;
        du[0] = (bf16_t)ru0.x; du[1] = (bf16_t)ru0.y; du[2] = (bf16_t)ru0.z; du[3] = (bf16_t)ru0.w;
        du[4] = (bf16_t)ru1.x; du[5] = (bf16_t)ru1.y; du[6] = (bf16_t)ru1.z; du[7] = (bf16_t)ru1.w;
    };

    constexpr int NS = Hh / BK;

    // prologue: stage tile 0 into buffer 0
    loadB(0);
    commitB(0);
    async_copy32(aDst0, aSrc);
    wait_async();
    __syncthreads();

    for (int step = 0; step < NS; ++step) {
        const int cur = step & 1, nxt = cur ^ 1;
        const bool more = (step + 1) < NS;
        if (more) {                      // issue next tile's global traffic first
            loadB((step + 1) * BK);
            async_copy32(nxt ? aDst1 : aDst0, aSrc + (step + 1) * BK);
        }
        // ---- compute from current buffer ----
        v16bf a[2], fbg[2], fbu[2];
#pragma unroll
        for (int mi = 0; mi < 2; mi++) {
            int row = wm * 32 + mi * 16 + lm;
            a[mi] = cat16(*(const v8bf*)&As[cur][row][kh * 16],
                          *(const v8bf*)&As[cur][row][kh * 16 + 8]);
        }
#pragma unroll
        for (int ni = 0; ni < 2; ni++) {
            int col = wn * 32 + ni * 16 + lm;
            fbg[ni] = cat16(*(const v8bf*)&Bg[cur][col][kh * 16],
                            *(const v8bf*)&Bg[cur][col][kh * 16 + 8]);
            fbu[ni] = cat16(*(const v8bf*)&Bu[cur][col][kh * 16],
                            *(const v8bf*)&Bu[cur][col][kh * 16 + 8]);
        }
#pragma unroll
        for (int mi = 0; mi < 2; mi++)
#pragma unroll
            for (int ni = 0; ni < 2; ni++) {
                cg[mi][ni] = __builtin_amdgcn_wmma_f32_16x16x32_bf16(
                    false, a[mi], false, fbg[ni], (short)0, cg[mi][ni], false, false);
                cu[mi][ni] = __builtin_amdgcn_wmma_f32_16x16x32_bf16(
                    false, a[mi], false, fbu[ni], (short)0, cu[mi][ni], false, false);
            }
        if (more) {
            commitB(nxt);
            wait_async();
        }
        __syncthreads();
    }

    // ---- fused SiLU epilogue -> bf16 act ----
#pragma unroll
    for (int mi = 0; mi < 2; mi++)
#pragma unroll
        for (int ni = 0; ni < 2; ni++) {
#pragma unroll
            for (int r = 0; r < 8; r++) {
                int ml = wm * 32 + mi * 16 + kh * 8 + r;
                int row = mbase + ml;
                if (row < nrows) {
                    int n = nbase + wn * 32 + ni * 16 + lm;
                    float g = cg[mi][ni][r], u = cu[mi][ni][r];
                    float act = (g / (1.0f + __expf(-g))) * u * rW[ml];
                    size_t orow = ROUTED ? ((size_t)e * Tt + row) : (size_t)row;
                    Out[orow * F + n] = (bf16_t)act;
                }
            }
        }
}

// GEMM2: Y(+)= act @ Wdown^T.  Shared: plain store (covers all of Y).
// Routed: scatter rows back to tokens via global f32 atomics.
template <bool ROUTED>
__global__ __launch_bounds__(256) void gemm_down_kernel(
    const bf16_t* __restrict__ Act,  // [(E*Tt | Tt), K] bf16
    const float* __restrict__ Wd,    // [(E*)H, K] fp32 (rows = output h)
    float* __restrict__ Y,           // [Tt, Hh]
    const int* __restrict__ cnt, const int* __restrict__ tok, const int K) {
    const int e = blockIdx.z;
    const int nrows = ROUTED ? cnt[e] : Tt;
    const int mbase = blockIdx.y * BM;
    if (mbase >= nrows) return;
    const int nbase = blockIdx.x * BN;
    const float* W = Wd + (ROUTED ? (size_t)e * Hh * K : 0);
    const bf16_t* A = Act + (ROUTED ? (size_t)e * Tt * K : 0);

    __shared__ alignas(16) bf16_t As[2][BM][LDK];
    __shared__ alignas(16) bf16_t Bs[2][BN][LDK];
    __shared__ int rTok[BM];

    const int tid = threadIdx.x;
    if (ROUTED && tid < BM) {
        int r = mbase + tid;
        rTok[tid] = (r < nrows) ? tok[e * Tt + r] : 0;
    }
    __syncthreads();

    const int wave = tid >> 5, lane = tid & 31;
    const int wm = wave & 3, wn = wave >> 2;
    const int lm = lane & 15, kh = lane >> 4;

    const int arow = tid >> 1, ah = (tid & 1) * 16;
    const int brow = tid >> 2, bq = (tid & 3) * 8;

    const bf16_t* aSrc = A + (size_t)(mbase + arow) * K + ah;
    const float*  bSrc = W + (size_t)(nbase + brow) * K + bq;
    const unsigned aDst0 = lds_off(&As[0][arow][ah]);
    const unsigned aDst1 = lds_off(&As[1][arow][ah]);

    v8f c[2][2] = {};
    float4 rb0, rb1;

    auto loadB = [&](int k0) {
        rb0 = *(const float4*)(bSrc + k0);
        rb1 = *(const float4*)(bSrc + k0 + 4);
    };
    auto commitB = [&](int buf) {
        bf16_t* db = &Bs[buf][brow][bq];
        db[0] = (bf16_t)rb0.x; db[1] = (bf16_t)rb0.y; db[2] = (bf16_t)rb0.z; db[3] = (bf16_t)rb0.w;
        db[4] = (bf16_t)rb1.x; db[5] = (bf16_t)rb1.y; db[6] = (bf16_t)rb1.z; db[7] = (bf16_t)rb1.w;
    };

    const int NS = K / BK;

    loadB(0);
    commitB(0);
    async_copy32(aDst0, aSrc);
    wait_async();
    __syncthreads();

    for (int step = 0; step < NS; ++step) {
        const int cur = step & 1, nxt = cur ^ 1;
        const bool more = (step + 1) < NS;
        if (more) {
            loadB((step + 1) * BK);
            async_copy32(nxt ? aDst1 : aDst0, aSrc + (step + 1) * BK);
        }
        v16bf a[2], b[2];
#pragma unroll
        for (int mi = 0; mi < 2; mi++) {
            int row = wm * 32 + mi * 16 + lm;
            a[mi] = cat16(*(const v8bf*)&As[cur][row][kh * 16],
                          *(const v8bf*)&As[cur][row][kh * 16 + 8]);
        }
#pragma unroll
        for (int ni = 0; ni < 2; ni++) {
            int col = wn * 32 + ni * 16 + lm;
            b[ni] = cat16(*(const v8bf*)&Bs[cur][col][kh * 16],
                          *(const v8bf*)&Bs[cur][col][kh * 16 + 8]);
        }
#pragma unroll
        for (int mi = 0; mi < 2; mi++)
#pragma unroll
            for (int ni = 0; ni < 2; ni++)
                c[mi][ni] = __builtin_amdgcn_wmma_f32_16x16x32_bf16(
                    false, a[mi], false, b[ni], (short)0, c[mi][ni], false, false);
        if (more) {
            commitB(nxt);
            wait_async();
        }
        __syncthreads();
    }

#pragma unroll
    for (int mi = 0; mi < 2; mi++)
#pragma unroll
        for (int ni = 0; ni < 2; ni++) {
#pragma unroll
            for (int r = 0; r < 8; r++) {
                int ml = wm * 32 + mi * 16 + kh * 8 + r;
                int row = mbase + ml;
                int n = nbase + wn * 32 + ni * 16 + lm;
                float v = c[mi][ni][r];
                if constexpr (ROUTED) {
                    if (row < nrows)
                        unsafeAtomicAdd(&Y[(size_t)rTok[ml] * Hh + n], v);
                } else {
                    Y[(size_t)row * Hh + n] = v;
                }
            }
        }
}

extern "C" void kernel_launch(void* const* d_in, const int* in_sizes, int n_in,
                              void* d_out, int out_size, void* d_ws, size_t ws_size,
                              hipStream_t stream) {
    const float* x   = (const float*)d_in[0];  // [1,2048,2048]
    const float* gw  = (const float*)d_in[1];  // [8,2048]
    const float* wgu = (const float*)d_in[2];  // [8,2816,2048]
    const float* wdn = (const float*)d_in[3];  // [8,2048,1408]
    const float* sgu = (const float*)d_in[4];  // [5632,2048]
    const float* sdn = (const float*)d_in[5];  // [2048,2816]
    float* y = (float*)d_out;

    char* ws = (char*)d_ws;
    size_t off = 0;
    auto take = [&](size_t bytes) -> char* {
        char* p = ws + off;
        off = (off + bytes + 255) & ~(size_t)255;
        return p;
    };
    bf16_t* xb   = (bf16_t*)take((size_t)Tt * Hh * sizeof(bf16_t));       // 8 MB
    bf16_t* sact = (bf16_t*)take((size_t)Tt * Fs * sizeof(bf16_t));       // 11.5 MB
    bf16_t* ract = (bf16_t*)take((size_t)Ee * Tt * Ff * sizeof(bf16_t));  // 46 MB (worst-case caps)
    int*    cnt  = (int*)take(Ee * sizeof(int));
    int*    tok  = (int*)take((size_t)Ee * Tt * sizeof(int));
    float*  wv   = (float*)take((size_t)Ee * Tt * sizeof(float));
    (void)in_sizes; (void)n_in; (void)out_size; (void)ws_size;

    zero_cnt_kernel<<<1, 32, 0, stream>>>(cnt);
    convert_x_kernel<<<(Tt * Hh / 4) / 256, 256, 0, stream>>>(x, xb);
    gate_topk_kernel<<<Tt / 8, 256, 0, stream>>>(x, gw, cnt, tok, wv);

    // shared expert: gu+silu  (M=2048, N=Fs, K=H)
    gemm_gu_act_kernel<false><<<dim3(Fs / BN, Tt / BM, 1), 256, 0, stream>>>(
        xb, sgu, sact, nullptr, nullptr, nullptr, Fs);
    // routed experts: gu+silu*gate  (per-expert grouped, early-exit past cnt[e])
    gemm_gu_act_kernel<true><<<dim3(Ff / BN, Tt / BM, Ee), 256, 0, stream>>>(
        xb, wgu, ract, cnt, tok, wv, Ff);

    // shared down: plain store covers all of Y
    gemm_down_kernel<false><<<dim3(Hh / BN, Tt / BM, 1), 256, 0, stream>>>(
        sact, sdn, y, nullptr, nullptr, Fs);
    // routed down: scatter-accumulate into Y
    gemm_down_kernel<true><<<dim3(Hh / BN, Tt / BM, Ee), 256, 0, stream>>>(
        ract, wdn, y, cnt, tok, Ff);
}
